// TensorProductConvLayer_38087769981432
// MI455X (gfx1250) — compile-verified, matrix operand order checked by reading, not run on hardware
//
#include <hip/hip_runtime.h>
#include <hip/hip_bf16.h>

#define NS 32
#define NV 8
#define N_NODES 16384
#define N_EDGES 131072
#define NEF 128
#define HIDN 128
#define WN 1600
#define DD 56            // NS + 3*NV
#define WSTR 1608        // padded LDS stride for w tiles (bank spread)

#define PATH_NORM 0.15811388300841897f   // 1/sqrt(40)
#define INV_SQRT3 0.57735026918962576f
#define EPSV 1e-5f

typedef __attribute__((ext_vector_type(16))) __bf16 v16bf;
typedef __attribute__((ext_vector_type(8)))  float  v8f;

// Native f32 -> bf16 conversion (RNE on gfx1250; lowers to v_cvt hardware op).
__device__ __forceinline__ __bf16 f2bf(float f) { return (__bf16)f; }

// A-fragment (16x32 bf16) from f32 row in LDS.
// Lane L<16: K = {0..7, 16..23}+kb ; L>=16: kb=8 shifts to {8..15, 24..31}.
__device__ __forceinline__ v16bf load_afrag_f32(const float* p) {
  v16bf a;
#pragma unroll
  for (int h = 0; h < 8; ++h) { a[h] = f2bf(p[h]); a[h + 8] = f2bf(p[h + 16]); }
  return a;
}

__device__ __forceinline__ v16bf load_afrag_bf(const __bf16* p) {
  v16bf a;
#pragma unroll
  for (int h = 0; h < 8; ++h) { a[h] = p[h]; a[h + 8] = p[h + 16]; }
  return a;
}

__device__ __forceinline__ v8f wmma_bf(v16bf a, v16bf b, v8f c) {
  return __builtin_amdgcn_wmma_f32_16x16x32_bf16(false, a, false, b, (short)0, c,
                                                 false, false);
}

// ---------------------------------------------------------------------------
// Kernel 0: zero scatter buffers + stats
// ---------------------------------------------------------------------------
__global__ void zero_ws(float* __restrict__ p, int nelem) {
  int idx = blockIdx.x * 256 + threadIdx.x;
  if (idx < nelem) p[idx] = 0.0f;
}

// ---------------------------------------------------------------------------
// Kernel 1: pack W1 (128x128) and W2 (128x1600) f32 -> bf16 WMMA B-fragment
// layout. Fragment (kt, nt): elem index ((kt*NT + nt)*32 + lane)*16 + h,
// k = 32*kt + kb + (h<8 ? h : h+8), kb = (lane<16 ? 0 : 8), n = 16*nt + lane%16
// ---------------------------------------------------------------------------
__global__ void prep_frags(const float* __restrict__ W1, const float* __restrict__ W2,
                           __bf16* __restrict__ w1f, __bf16* __restrict__ w2f) {
  int o = blockIdx.x * 256 + threadIdx.x;
  if (o >= 16384 + 204800) return;
  if (o < 16384) {
    int h = o & 15, lane = (o >> 4) & 31, rest = o >> 9;
    int nt = rest & 7, kt = rest >> 3;
    int kb = (lane < 16) ? 0 : 8;
    int k = kt * 32 + kb + ((h < 8) ? h : h + 8);
    int n = nt * 16 + (lane & 15);
    w1f[o] = f2bf(W1[k * HIDN + n]);
  } else {
    int o2 = o - 16384;
    int h = o2 & 15, lane = (o2 >> 4) & 31, rest = o2 >> 9;
    int nt = rest % 100, kt = rest / 100;
    int kb = (lane < 16) ? 0 : 8;
    int k = kt * 32 + kb + ((h < 8) ? h : h + 8);
    int n = nt * 16 + (lane & 15);
    w2f[o2] = f2bf(W2[k * WN + n]);
  }
}

// ---------------------------------------------------------------------------
// Kernel 2: fused per-edge MLP (WMMA bf16) + tensor product + atomic scatter.
// One block = 32 edges, 8 waves, dynamic LDS.
// ---------------------------------------------------------------------------
__global__ __launch_bounds__(256, 1)
void edge_kernel(const float* __restrict__ node_attr,
                 const float* __restrict__ edge_attr,
                 const float* __restrict__ edge_sh,
                 const float* __restrict__ b1g,
                 const float* __restrict__ b2g,
                 const __bf16* __restrict__ w1f,
                 const __bf16* __restrict__ w2f,
                 const long long* __restrict__ eidx,
                 float* __restrict__ sums,
                 float* __restrict__ cnt) {
  extern __shared__ char smem[];
  float*  w_lds  = (float*)smem;                    // 32*WSTR f32
  float*  ea_lds = w_lds + 32 * WSTR;               // 32*128 f32
  float*  x_lds  = ea_lds + 32 * NEF;               // 32*56 f32
  float*  sh_lds = x_lds + 32 * DD;                 // 32*4 f32
  float*  dv_lds = sh_lds + 32 * 4;                 // 32*8 f32
  __bf16* h_lds  = (__bf16*)(dv_lds + 32 * 8);      // 32*128 bf16
  int*    src_lds = (int*)(h_lds + 32 * HIDN);      // 32
  int*    dst_lds = src_lds + 32;                   // 32

  const int tid = threadIdx.x;
  const int lane = tid & 31;
  const int wv = tid >> 5;
  const long long e0 = (long long)blockIdx.x * 32;

  if (tid < 32)      src_lds[tid]      = (int)eidx[e0 + tid];
  else if (tid < 64) dst_lds[tid - 32] = (int)eidx[(long long)N_EDGES + e0 + (tid - 32)];
#pragma unroll
  for (int i = tid; i < 32 * NEF; i += 256) ea_lds[i] = edge_attr[e0 * NEF + i];
  if (tid < 128) sh_lds[tid] = edge_sh[e0 * 4 + tid];
  __syncthreads();
  for (int i = tid; i < 32 * DD; i += 256) {
    int e = i / DD, f = i - e * DD;
    x_lds[i] = node_attr[(long long)dst_lds[e] * DD + f];
  }
  __syncthreads();

  // ---- Phase 1: h = relu(ea @ W1 + b1) -> bf16 in LDS ----
  {
    const int mt = wv >> 2;               // waves 0-3: edge tile 0, 4-7: tile 1
    const int m = lane & 15;
    const int kb = (lane < 16) ? 0 : 8;
    const float* row = ea_lds + (mt * 16 + m) * NEF;
    v16bf a0 = load_afrag_f32(row + 0  + kb);
    v16bf a1 = load_afrag_f32(row + 32 + kb);
    v16bf a2 = load_afrag_f32(row + 64 + kb);
    v16bf a3 = load_afrag_f32(row + 96 + kb);
#pragma unroll
    for (int s = 0; s < 2; ++s) {
      const int nt = ((wv & 3) << 1) | s;
      const float bias = b1g[nt * 16 + m];
      v8f c;
#pragma unroll
      for (int r = 0; r < 8; ++r) c[r] = bias;
      const __bf16* bp = w1f + (nt * 32 + lane) * 16;   // kt stride = 8*32*16
      c = wmma_bf(a0, *(const v16bf*)(bp + 0),     c);
      c = wmma_bf(a1, *(const v16bf*)(bp + 4096),  c);
      c = wmma_bf(a2, *(const v16bf*)(bp + 8192),  c);
      c = wmma_bf(a3, *(const v16bf*)(bp + 12288), c);
      const int mbase = mt * 16 + ((lane >> 4) << 3);
      const int n = nt * 16 + m;
#pragma unroll
      for (int r = 0; r < 8; ++r)
        h_lds[(mbase + r) * HIDN + n] = f2bf(fmaxf(c[r], 0.0f));
    }
  }
  // dot_vv precompute (one (e,i) pair per thread)
  {
    const int e = tid >> 3, i = tid & 7;
    const float* xv = x_lds + e * DD + NS + i * 3;
    const float* sh = sh_lds + e * 4;
    dv_lds[e * 8 + i] = INV_SQRT3 * (xv[0] * sh[1] + xv[1] * sh[2] + xv[2] * sh[3]);
  }
  __syncthreads();

  // ---- Phase 2: w = h @ W2 + b2 -> f32 in LDS (32 x 1600, stride WSTR) ----
  {
    const int m = lane & 15;
    const int kb = (lane < 16) ? 0 : 8;
#pragma unroll 1
    for (int half = 0; half < 2; ++half) {
      const __bf16* hrow = h_lds + (half * 16 + m) * HIDN + kb;
      v16bf a0 = load_afrag_bf(hrow + 0);
      v16bf a1 = load_afrag_bf(hrow + 32);
      v16bf a2 = load_afrag_bf(hrow + 64);
      v16bf a3 = load_afrag_bf(hrow + 96);
#pragma unroll 1
      for (int nt = wv; nt < 100; nt += 8) {
        const float bias = b2g[nt * 16 + m];
        v8f c;
#pragma unroll
        for (int r = 0; r < 8; ++r) c[r] = bias;
        const __bf16* bp = w2f + (nt * 32 + lane) * 16;  // kt stride = 100*32*16
        c = wmma_bf(a0, *(const v16bf*)(bp + 0),      c);
        c = wmma_bf(a1, *(const v16bf*)(bp + 51200),  c);
        c = wmma_bf(a2, *(const v16bf*)(bp + 102400), c);
        c = wmma_bf(a3, *(const v16bf*)(bp + 153600), c);
        const int mbase = half * 16 + ((lane >> 4) << 3);
        const int n = nt * 16 + m;
#pragma unroll
        for (int r = 0; r < 8; ++r)
          w_lds[(mbase + r) * WSTR + n] = c[r];
      }
    }
  }
  __syncthreads();

  // ---- Phase 3: tensor product + scatter (8 threads per edge) ----
  {
    const int e = tid >> 3;
    const int t = tid & 7;
    const float* w = w_lds + e * WSTR;
    const float* x = x_lds + e * DD;
    const float* dv = dv_lds + e * 8;
    const float ss  = sh_lds[e * 4 + 0];
    const float sv0 = sh_lds[e * 4 + 1];
    const float sv1 = sh_lds[e * 4 + 2];
    const float sv2 = sh_lds[e * 4 + 3];
    float os[4];
#pragma unroll
    for (int jj = 0; jj < 4; ++jj) {
      const int j = t * 4 + jj;
      float a = 0.f, b = 0.f;
#pragma unroll
      for (int i = 0; i < NS; ++i) a += w[i * NS + j] * x[i];
#pragma unroll
      for (int i = 0; i < NV; ++i) b += w[1024 + i * NS + j] * dv[i];
      os[jj] = (a * ss + b) * PATH_NORM;
    }
    float asv = 0.f;
#pragma unroll
    for (int i = 0; i < NS; ++i) asv += w[1280 + i * NV + t] * x[i];
    float av0 = 0.f, av1 = 0.f, av2 = 0.f;
#pragma unroll
    for (int i = 0; i < NV; ++i) {
      const float wvv = w[1536 + i * NV + t];
      av0 += wvv * x[NS + i * 3 + 0];
      av1 += wvv * x[NS + i * 3 + 1];
      av2 += wvv * x[NS + i * 3 + 2];
    }
    const float ov0 = (asv * sv0 + av0 * ss) * PATH_NORM;
    const float ov1 = (asv * sv1 + av1 * ss) * PATH_NORM;
    const float ov2 = (asv * sv2 + av2 * ss) * PATH_NORM;
    float* dstp = sums + (long long)src_lds[e] * DD;
#pragma unroll
    for (int jj = 0; jj < 4; ++jj) atomicAdd(dstp + t * 4 + jj, os[jj]);
    atomicAdd(dstp + NS + t * 3 + 0, ov0);
    atomicAdd(dstp + NS + t * 3 + 1, ov1);
    atomicAdd(dstp + NS + t * 3 + 2, ov2);
    if (t == 0) atomicAdd(cnt + src_lds[e], 1.0f);
  }
}

// ---------------------------------------------------------------------------
// Kernel 3: mean + residual (in place) + feature statistics
// stats: [0:32) sum(s), [32:64) sum(s^2), [64:72) sum(|v_j|^2)
// ---------------------------------------------------------------------------
__global__ void node_combine(float* __restrict__ buf, const float* __restrict__ cnt,
                             const float* __restrict__ node_attr,
                             float* __restrict__ stats) {
  __shared__ float ls[72];
  const int tid = threadIdx.x;
  if (tid < 72) ls[tid] = 0.0f;
  __syncthreads();
  const long long n = (long long)blockIdx.x * 256 + tid;
  const float inv = 1.0f / fmaxf(cnt[n], 1.0f);
  float out[DD];
#pragma unroll
  for (int f = 0; f < DD; ++f) {
    out[f] = buf[n * DD + f] * inv + node_attr[n * DD + f];
    buf[n * DD + f] = out[f];
  }
#pragma unroll
  for (int f = 0; f < NS; ++f) {
    atomicAdd(&ls[f], out[f]);
    atomicAdd(&ls[NS + f], out[f] * out[f]);
  }
#pragma unroll
  for (int j = 0; j < NV; ++j) {
    const float* v = out + NS + j * 3;
    atomicAdd(&ls[64 + j], v[0] * v[0] + v[1] * v[1] + v[2] * v[2]);
  }
  __syncthreads();
  if (tid < 72) atomicAdd(&stats[tid], ls[tid]);
}

// ---------------------------------------------------------------------------
// Kernel 4: finalize normalization constants
// finals: [0:32) sscale, [32:64) sshift, [64:72) vscale
// ---------------------------------------------------------------------------
__global__ void finalize_stats(const float* __restrict__ stats,
                               const float* __restrict__ gamma,
                               const float* __restrict__ beta,
                               float* __restrict__ finals) {
  const int t = threadIdx.x;
  const float invN = 1.0f / (float)N_NODES;
  if (t < NS) {
    const float mean = stats[t] * invN;
    const float var = stats[NS + t] * invN - mean * mean;
    const float sc = gamma[t] / sqrtf(var + EPSV);
    finals[t] = sc;
    finals[NS + t] = beta[t] - mean * sc;
  } else if (t < NS + NV) {
    const int j = t - NS;
    const float vn2 = stats[64 + j] * invN;
    finals[64 + j] = gamma[NS + j] / sqrtf(vn2 + EPSV);
  }
}

// ---------------------------------------------------------------------------
// Kernel 5: apply normalization -> d_out
// ---------------------------------------------------------------------------
__global__ void apply_norm(const float* __restrict__ buf,
                           const float* __restrict__ finals,
                           float* __restrict__ out) {
  const int idx = blockIdx.x * 256 + threadIdx.x;
  if (idx >= N_NODES * DD) return;
  const int f = idx % DD;
  const float v = buf[idx];
  out[idx] = (f < NS) ? (v * finals[f] + finals[NS + f])
                      : (v * finals[64 + (f - NS) / 3]);
}

// ---------------------------------------------------------------------------
extern "C" void kernel_launch(void* const* d_in, const int* in_sizes, int n_in,
                              void* d_out, int out_size, void* d_ws, size_t ws_size,
                              hipStream_t stream) {
  const float* node_attr = (const float*)d_in[0];
  const float* edge_attr = (const float*)d_in[1];
  const float* edge_sh   = (const float*)d_in[2];
  const float* W1        = (const float*)d_in[3];
  const float* b1        = (const float*)d_in[4];
  const float* W2        = (const float*)d_in[5];
  const float* b2        = (const float*)d_in[6];
  const float* gamma     = (const float*)d_in[7];
  const float* beta      = (const float*)d_in[8];
  const long long* eidx  = (const long long*)d_in[9];

  float* ws     = (float*)d_ws;
  float* buf    = ws;                      // N_NODES*56 (sums, then out in place)
  float* cnt    = ws + 917504;             // N_NODES
  float* stats  = ws + 933888;             // 72
  float* finals = ws + 933960;             // 72
  __bf16* w1f   = (__bf16*)(ws + 934032);  // 16384 bf16
  __bf16* w2f   = w1f + 16384;             // 204800 bf16

  const int nz = 917504 + 16384 + 72;
  zero_ws<<<(nz + 255) / 256, 256, 0, stream>>>(ws, nz);
  prep_frags<<<(16384 + 204800 + 255) / 256, 256, 0, stream>>>(W1, W2, w1f, w2f);

  const size_t lds_bytes =
      (size_t)32 * WSTR * 4 + 32 * NEF * 4 + 32 * DD * 4 + 32 * 4 * 4 +
      32 * 8 * 4 + (size_t)32 * HIDN * 2 + 64 * 4;
  edge_kernel<<<N_EDGES / 32, 256, lds_bytes, stream>>>(
      node_attr, edge_attr, edge_sh, b1, b2, w1f, w2f, eidx, buf, cnt);

  node_combine<<<N_NODES / 256, 256, 0, stream>>>(buf, cnt, node_attr, stats);
  finalize_stats<<<1, 64, 0, stream>>>(stats, gamma, beta, finals);
  apply_norm<<<(N_NODES * DD + 255) / 256, 256, 0, stream>>>(buf, finals, (float*)d_out);
}
